// EncoderBlock_85169201480342
// MI455X (gfx1250) — matrix operand order repaired and checked by reading, not compiled, not run on hardware
//
#include <hip/hip_runtime.h>
#include <hip/hip_bf16.h>

// ---------------------------------------------------------------------------
// CDNA5 / gfx1250 transformer encoder block, v2.
//  * all contractions: v_wmma_f32_16x16x32_f16 (f16 in, f32 accumulate)
//  * all WMMA operands live in f16 in the workspace (epilogues write f16,
//    weights pre-transposed+converted once per launch)
//  * GEMM tiles staged with GLOBAL_LOAD_ASYNC_TO_LDS_B128 (ASYNCcnt),
//    double-buffered LDS, s_wait_asynccnt pipelining
//  * wave32, 256-thread blocks (8 waves)
// Workspace: MD*4 (x2 fp32) + (5*MD + 6.49M weights) * 2 bytes  ~ 57 MB.
// ---------------------------------------------------------------------------

typedef _Float16 f16;
typedef __attribute__((ext_vector_type(2)))  _Float16 f16x2;
typedef __attribute__((ext_vector_type(16))) _Float16 v16h;
typedef __attribute__((ext_vector_type(8)))  float    v8f;

#define WMMA_F16(a, b, c) \
  __builtin_amdgcn_wmma_f32_16x16x32_f16(false, (a), false, (b), (short)0, (c), false, false)

// Async copy of one 16-byte chunk per lane: global -> LDS (ASYNCcnt-tracked).
__device__ __forceinline__ unsigned lds_off(const void* p) {
  // flat shared pointer: addr[31:0] is the LDS byte address (ISA 10.2 aperture calc)
  return (unsigned)(unsigned long long)p;
}
__device__ __forceinline__ void async_copy_b128(const void* lds_dst, const void* gsrc) {
  asm volatile("global_load_async_to_lds_b128 %0, %1, off"
               :: "v"(lds_off(lds_dst)), "v"((unsigned long long)gsrc)
               : "memory");
}
#define WAIT_ASYNC(n) asm volatile("s_wait_asynccnt %0" :: "n"(n) : "memory")

// A-fragment: lane holds row `row`; element e maps to k = 16*(e>>3) + 8*hf + (e&7).
__device__ __forceinline__ v16h frag_a(const f16* base, int row, int stride, int hf) {
  union { v16h v; f16x2 p[8]; } u;
  const f16* r = base + row * stride + 8 * hf;
#pragma unroll
  for (int i = 0; i < 4; ++i) {
    u.p[i]     = *(const f16x2*)(r + 2 * i);
    u.p[i + 4] = *(const f16x2*)(r + 16 + 2 * i);
  }
  return u.v;
}
// B-fragment from a transposed LDS tile Bt[n][k]: lane holds column `col`,
// element e maps to k = 16*(e>>3) + 8*hf + (e&7) (same order as frag_a).
__device__ __forceinline__ v16h frag_b(const f16* base, int col, int stride, int hf) {
  union { v16h v; f16x2 p[8]; } u;
  const f16* r = base + col * stride + 8 * hf;
#pragma unroll
  for (int i = 0; i < 4; ++i) {
    u.p[i]     = *(const f16x2*)(r + 2 * i);
    u.p[i + 4] = *(const f16x2*)(r + 16 + 2 * i);
  }
  return u.v;
}

// ---------------------------------------------------------------------------
// Weight transpose+convert: Wt[n][k] (f16, row-major [N,K]) from fp32 source.
// Source addressing (head-blocked Wq/Wv = [H][D][HS] or plain [K,N]):
//   in(k,n) = W[(n/hw)*hstride + k*hw + (n%hw)]
// ---------------------------------------------------------------------------
__global__ __launch_bounds__(256) void wtrans_kernel(const float* __restrict__ W,
                                                     f16* __restrict__ Wt,
                                                     int N, int K, int hw,
                                                     size_t hstride) {
  __shared__ f16 T[64 * 72];
  const int tid = threadIdx.x;
  const int k0 = blockIdx.x * 64, n0 = blockIdx.y * 64;
  for (int i = tid; i < 64 * 64; i += 256) {
    int kk = i >> 6, nn = i & 63;                       // coalesced over n
    int n = n0 + nn, head = n / hw, c = n % hw;
    T[nn * 72 + kk] = (f16)W[head * hstride + (size_t)(k0 + kk) * hw + c];
  }
  __syncthreads();
  for (int i = tid; i < 64 * 64; i += 256) {
    int nn = i >> 6, kk = i & 63;                       // coalesced over k
    Wt[(size_t)(n0 + nn) * K + (k0 + kk)] = T[nn * 72 + kk];
  }
}

// ---------------------------------------------------------------------------
// LayerNorm over D=768, one block per row, f16 output (WMMA A operand).
// ---------------------------------------------------------------------------
__global__ __launch_bounds__(256) void ln_kernel(const float* __restrict__ x,
                                                 const float* __restrict__ gw,
                                                 const float* __restrict__ bw,
                                                 f16* __restrict__ out) {
  const int D = 768;
  const int row = blockIdx.x;
  const float* xr = x + (size_t)row * D;
  __shared__ float r1[8], r2[8];
  const int tid = threadIdx.x, lane = tid & 31, wv = tid >> 5;

  float s = 0.f;
  for (int i = tid; i < D; i += 256) s += xr[i];
#pragma unroll
  for (int off = 16; off; off >>= 1) s += __shfl_xor(s, off, 32);
  if (lane == 0) r1[wv] = s;
  __syncthreads();
  float mu = 0.f;
#pragma unroll
  for (int i = 0; i < 8; ++i) mu += r1[i];
  mu *= (1.f / 768.f);

  float vs = 0.f;
  for (int i = tid; i < D; i += 256) { float d = xr[i] - mu; vs += d * d; }
#pragma unroll
  for (int off = 16; off; off >>= 1) vs += __shfl_xor(vs, off, 32);
  if (lane == 0) r2[wv] = vs;
  __syncthreads();
  float var = 0.f;
#pragma unroll
  for (int i = 0; i < 8; ++i) var += r2[i];
  const float rstd = rsqrtf(var * (1.f / 768.f) + 1e-5f);

  for (int i = tid; i < D; i += 256)
    out[(size_t)row * D + i] = (f16)((xr[i] - mu) * rstd * gw[i] + bw[i]);
}

// ---------------------------------------------------------------------------
// WMMA GEMM with async double-buffered staging.
//   out = epilogue(A[M,K]_f16 @ Wt[N,K]_f16^T + bias [+resid_f32])
// Block tile 128x128, 8 waves (4x2), wave tile 32x64, K-step 32.
// ACT: 0 none, 1 exact GELU.  OUT16: write f16*oscale, else f32.
// ---------------------------------------------------------------------------
constexpr int ASTR = 56, WSTR = 56;   // f16 strides: 112B = mult of 16, conflict-free

__device__ __forceinline__ void gemm_issue(const f16* __restrict__ A,
                                           const f16* __restrict__ Wt, int K,
                                           int m0, int n0, int k0,
                                           f16* As, f16* Ws, int tid) {
#pragma unroll
  for (int it = 0; it < 2; ++it) {     // 512 chunks each of A and W, 2/thread
    int c = tid + it * 256;
    int r = c >> 2;                    // 0..127
    int kc = (c & 3) * 8;              // 0,8,16,24 (f16) = 16B chunks
    async_copy_b128(As + r * ASTR + kc, A  + (size_t)(m0 + r) * K + (k0 + kc));
    async_copy_b128(Ws + r * WSTR + kc, Wt + (size_t)(n0 + r) * K + (k0 + kc));
  }
}

template <int ACT, int OUT16, bool RES>
__global__ __launch_bounds__(256) void gemm_kernel(
    const f16* __restrict__ A, const f16* __restrict__ Wt,
    const float* __restrict__ bias, const float* __restrict__ resid,
    float* __restrict__ outf, f16* __restrict__ outh,
    int M, int N, int K, float oscale) {
  __shared__ __attribute__((aligned(16))) f16 As[2][128 * ASTR];
  __shared__ __attribute__((aligned(16))) f16 Ws[2][128 * WSTR];

  const int tid = threadIdx.x, lane = tid & 31, wv = tid >> 5;
  const int l16 = lane & 15, hf = lane >> 4;
  const int m0 = blockIdx.y * 128, n0 = blockIdx.x * 128;
  const int mw = (wv & 3) * 32, nw = (wv >> 2) * 64;

  v8f acc[2][4] = {};
  const int KT = K >> 5;

  gemm_issue(A, Wt, K, m0, n0, 0, As[0], Ws[0], tid);   // 4 async instr / wave
  for (int kb = 0; kb < KT; ++kb) {
    const int cur = kb & 1;
    if (kb + 1 < KT) {
      gemm_issue(A, Wt, K, m0, n0, (kb + 1) << 5, As[cur ^ 1], Ws[cur ^ 1], tid);
      WAIT_ASYNC(4);                   // leave only the prefetched tile in flight
    } else {
      WAIT_ASYNC(0);
    }
    __syncthreads();

    const f16* Ab = As[cur];
    const f16* Wb = Ws[cur];
    v16h a0 = frag_a(Ab, mw + l16, ASTR, hf);
    v16h a1 = frag_a(Ab, mw + 16 + l16, ASTR, hf);
#pragma unroll
    for (int t = 0; t < 4; ++t) {
      v16h bt = frag_b(Wb, nw + t * 16 + l16, WSTR, hf);
      acc[0][t] = WMMA_F16(a0, bt, acc[0][t]);
      acc[1][t] = WMMA_F16(a1, bt, acc[1][t]);
    }
    __syncthreads();                   // done reading before next overwrite
  }

#pragma unroll
  for (int i = 0; i < 2; ++i)
#pragma unroll
    for (int t = 0; t < 4; ++t) {
      const int col = n0 + nw + t * 16 + l16;
      const float bj = bias[col];
#pragma unroll
      for (int r = 0; r < 8; ++r) {
        const int row = m0 + mw + i * 16 + r + 8 * hf;  // C layout: M = vgpr + 8*half
        float v = acc[i][t][r] + bj;
        if (ACT == 1) v = 0.5f * v * (1.f + erff(v * 0.70710678118654752f));
        if (RES) v += resid[(size_t)row * N + col];
        if (OUT16) outh[(size_t)row * N + col] = (f16)(v * oscale);
        else       outf[(size_t)row * N + col] = v;
      }
    }
}

// ---------------------------------------------------------------------------
// Flash-style attention. Q16 holds sqrt(768^-0.5)-scaled f16 Q; K = Q (bug
// preserved), so scores get the full 768^-0.5 scale. Block = (b, head,
// 128-row q block); wave owns 16 query rows; 64-row key blocks streamed with
// async Q/K staging; online softmax via 16-lane shfl matching the C-layout;
// P -> per-wave LDS -> A-fragments -> P@V.
// ---------------------------------------------------------------------------
__global__ __launch_bounds__(256) void attn_kernel(const f16* __restrict__ Q16,
                                                   const f16* __restrict__ V16,
                                                   f16* __restrict__ outh) {
  const int S = 512, D = 768, HS = 64;
  constexpr int QSTR = 72, KSTR = 72, PSTR = 72;   // 144B: mult of 16, conflict-free
  __shared__ __attribute__((aligned(16))) f16 Qs[128 * QSTR];
  __shared__ __attribute__((aligned(16))) f16 Ks[64 * KSTR];
  __shared__ __attribute__((aligned(16))) f16 Vs[64 * KSTR];   // Vs[n][k] transposed
  __shared__ __attribute__((aligned(16))) f16 Ps[8][16 * PSTR];

  const int bid = blockIdx.x;
  const int qb = bid & 3;
  const int h  = (bid >> 2) % 12;
  const int b  = bid / 48;
  const int tid = threadIdx.x, lane = tid & 31, wv = tid >> 5;
  const int l16 = lane & 15, hf = lane >> 4;
  const int qbase = qb * 128;

  const f16* Qg = Q16 + (size_t)b * S * D + h * HS;
  const f16* Vg = V16 + (size_t)b * S * D + h * HS;

  // async-stage the 128x64 query tile (4 instr/wave)
#pragma unroll
  for (int it = 0; it < 4; ++it) {
    int c = tid + it * 256;            // 1024 chunks
    int r = c >> 3, kc = (c & 7) * 8;
    async_copy_b128(Qs + r * QSTR + kc, Qg + (size_t)(qbase + r) * D + kc);
  }

  v8f oacc[4] = {};
  float mrow[8], lrow[8];
#pragma unroll
  for (int r = 0; r < 8; ++r) { mrow[r] = -1e30f; lrow[r] = 0.f; }

  for (int j = 0; j < 8; ++j) {
    __syncthreads();                   // previous Ks/Vs fully consumed
    // async-stage K block (= Q rows, bug preserved): 2 instr/wave
#pragma unroll
    for (int it = 0; it < 2; ++it) {
      int c = tid + it * 256;          // 512 chunks
      int r = c >> 3, kc = (c & 7) * 8;
      async_copy_b128(Ks + r * KSTR + kc, Qg + (size_t)(j * 64 + r) * D + kc);
    }
    // manual transposed stage of V block: Vs[n][k] = V[j*64+k][n]
#pragma unroll
    for (int it = 0; it < 8; ++it) {
      int i = tid + it * 256;          // 2048 f16x2 chunks
      int r = i >> 5, c2 = (i & 31) * 2;
      f16x2 v = *(const f16x2*)(Vg + (size_t)(j * 64 + r) * D + c2);
      Vs[c2 * KSTR + r] = v[0];
      Vs[(c2 + 1) * KSTR + r] = v[1];
    }
    WAIT_ASYNC(0);
    __syncthreads();

    const v16h aQ0 = frag_a(Qs,      16 * wv + l16, QSTR, hf);   // k 0..31
    const v16h aQ1 = frag_a(Qs + 32, 16 * wv + l16, QSTR, hf);   // k 32..63

    // scores: 16(q) x 64(kj) panel per wave
    v8f sacc[4] = {};
#pragma unroll
    for (int t = 0; t < 4; ++t) {
      v16h b0 = frag_b(Ks,      t * 16 + l16, KSTR, hf);
      v16h b1 = frag_b(Ks + 32, t * 16 + l16, KSTR, hf);
      sacc[t] = WMMA_F16(aQ0, b0, sacc[t]);
      sacc[t] = WMMA_F16(aQ1, b1, sacc[t]);
    }

    // online softmax per accumulator row r (row = r + 8*hf)
#pragma unroll
    for (int r = 0; r < 8; ++r) {
      float mx = fmaxf(fmaxf(sacc[0][r], sacc[1][r]), fmaxf(sacc[2][r], sacc[3][r]));
      mx = fmaxf(mx, __shfl_xor(mx, 1, 32));
      mx = fmaxf(mx, __shfl_xor(mx, 2, 32));
      mx = fmaxf(mx, __shfl_xor(mx, 4, 32));
      mx = fmaxf(mx, __shfl_xor(mx, 8, 32));
      const float mnew  = fmaxf(mrow[r], mx);
      const float alpha = __expf(mrow[r] - mnew);
      float rs = 0.f;
#pragma unroll
      for (int t = 0; t < 4; ++t) {
        float p = __expf(sacc[t][r] - mnew);
        sacc[t][r] = p;
        rs += p;
      }
      rs += __shfl_xor(rs, 1, 32);
      rs += __shfl_xor(rs, 2, 32);
      rs += __shfl_xor(rs, 4, 32);
      rs += __shfl_xor(rs, 8, 32);
      lrow[r] = lrow[r] * alpha + rs;
      mrow[r] = mnew;
#pragma unroll
      for (int t = 0; t < 4; ++t) oacc[t][r] *= alpha;
    }

    // P: C-layout -> per-wave LDS -> A-fragment layout (wave-private)
    f16* Pw = Ps[wv];
#pragma unroll
    for (int t = 0; t < 4; ++t)
#pragma unroll
      for (int r = 0; r < 8; ++r)
        Pw[(r + 8 * hf) * PSTR + t * 16 + l16] = (f16)sacc[t][r];

    const v16h aP0 = frag_a(Pw,      l16, PSTR, hf);
    const v16h aP1 = frag_a(Pw + 32, l16, PSTR, hf);
#pragma unroll
    for (int t = 0; t < 4; ++t) {
      v16h b0 = frag_b(Vs,      t * 16 + l16, KSTR, hf);
      v16h b1 = frag_b(Vs + 32, t * 16 + l16, KSTR, hf);
      oacc[t] = WMMA_F16(aP0, b0, oacc[t]);
      oacc[t] = WMMA_F16(aP1, b1, oacc[t]);
    }
  }

  // normalize + store f16 into concat layout [B,S,H*HS]
  f16* Og = outh + (size_t)b * S * D + h * HS;
#pragma unroll
  for (int t = 0; t < 4; ++t)
#pragma unroll
    for (int r = 0; r < 8; ++r) {
      const int row = qbase + 16 * wv + r + 8 * hf;
      Og[(size_t)row * D + t * 16 + l16] = (f16)(oacc[t][r] / lrow[r]);
    }
}

// ---------------------------------------------------------------------------
extern "C" void kernel_launch(void* const* d_in, const int* in_sizes, int n_in,
                              void* d_out, int out_size, void* d_ws, size_t ws_size,
                              hipStream_t stream) {
  (void)in_sizes; (void)n_in; (void)out_size; (void)ws_size;

  const float* x    = (const float*)d_in[0];
  const float* ln1g = (const float*)d_in[1];
  const float* ln1b = (const float*)d_in[2];
  const float* Wq   = (const float*)d_in[3];
  const float* bq   = (const float*)d_in[4];
  const float* Wv   = (const float*)d_in[5];
  const float* bv   = (const float*)d_in[6];
  const float* Wo   = (const float*)d_in[7];
  const float* bo   = (const float*)d_in[8];
  const float* ln2g = (const float*)d_in[9];
  const float* ln2b = (const float*)d_in[10];
  const float* W1   = (const float*)d_in[11];
  const float* b1   = (const float*)d_in[12];
  const float* W2   = (const float*)d_in[13];
  const float* b2   = (const float*)d_in[14];
  float* out = (float*)d_out;

  const int M = 4096, D = 768, F = 3072;
  const size_t MD = (size_t)M * D;
  const size_t WDD = (size_t)D * D;      // 589824
  const size_t WDF = (size_t)D * F;      // 2359296

  float* x2   = (float*)d_ws;            // [M,D] f32 (residual + LN2 input)
  f16* h16    = (f16*)(x2 + MD);         // LN1 out           [M,D]
  f16* Q16    = h16 + MD;                // sqrt-scaled Q+bq  [M,D] ([B,S,H*HS])
  f16* V16    = Q16 + MD;                // V+bv              [M,D]
  f16* cc16   = V16 + MD;                // attention concat  [M,D]
  f16* h2_16  = cc16 + MD;               // LN2 out           [M,D]
  f16* Wq_t   = h2_16 + MD;              // [768,768]
  f16* Wv_t   = Wq_t + WDD;
  f16* Wo_t   = Wv_t + WDD;
  f16* W1_t   = Wo_t + WDD;              // [3072,768]
  f16* W2_t   = W1_t + WDF;              // [768,3072]
  f16* g16    = h16;                     // FFN mid [M,F] reuses h16..cc16 (4*MD)

  const dim3 blk(256);
  const float sqs = 0.18995892f;         // sqrt(768^-0.5); Q*K gets 768^-0.5

  // one-time weight transpose+convert to f16 [N,K]
  wtrans_kernel<<<dim3(12, 12), blk, 0, stream>>>(Wq, Wq_t, D, D, 64, (size_t)D * 64);
  wtrans_kernel<<<dim3(12, 12), blk, 0, stream>>>(Wv, Wv_t, D, D, 64, (size_t)D * 64);
  wtrans_kernel<<<dim3(12, 12), blk, 0, stream>>>(Wo, Wo_t, D, D, D, 0);
  wtrans_kernel<<<dim3(12, 48), blk, 0, stream>>>(W1, W1_t, F, D, F, 0);
  wtrans_kernel<<<dim3(48, 12), blk, 0, stream>>>(W2, W2_t, D, F, D, 0);

  ln_kernel<<<M, blk, 0, stream>>>(x, ln1g, ln1b, h16);

  // Q (sqrt-scale folded) and V projections -> f16
  gemm_kernel<0, 1, false><<<dim3(D / 128, M / 128), blk, 0, stream>>>(
      h16, Wq_t, bq, nullptr, nullptr, Q16, M, D, D, sqs);
  gemm_kernel<0, 1, false><<<dim3(D / 128, M / 128), blk, 0, stream>>>(
      h16, Wv_t, bv, nullptr, nullptr, V16, M, D, D, 1.f);

  attn_kernel<<<8 * 12 * 4, blk, 0, stream>>>(Q16, V16, cc16);

  // out projection + residual(x) -> x2 (f32)
  gemm_kernel<0, 0, true><<<dim3(D / 128, M / 128), blk, 0, stream>>>(
      cc16, Wo_t, bo, x, x2, nullptr, M, D, D, 1.f);

  ln_kernel<<<M, blk, 0, stream>>>(x2, ln2g, ln2b, h2_16);

  // FFN1 + exact GELU -> f16
  gemm_kernel<1, 1, false><<<dim3(F / 128, M / 128), blk, 0, stream>>>(
      h2_16, W1_t, b1, nullptr, nullptr, g16, M, F, D, 1.f);
  // FFN2 + residual(x2) -> final f32 output
  gemm_kernel<0, 0, true><<<dim3(D / 128, M / 128), blk, 0, stream>>>(
      g16, W2_t, b2, x2, out, nullptr, M, D, F, 1.f);
}
